// RotatedTaskAlignedAssigner_14774687498824
// MI455X (gfx1250) — matrix operand order repaired and checked by reading, not compiled, not run on hardware
//
#include <hip/hip_runtime.h>
#include <stdint.h>

typedef _Float16 v16h __attribute__((ext_vector_type(16)));
typedef float    v8f  __attribute__((ext_vector_type(8)));

#define TOPK_K   13
#define IOU_EPSF 1e-7f
#define NORM_EPS 1e-9f

// ---------------------------------------------------------------------------
// K0a: per-GT precompute: gaussian params + spatial-test params
// gtp[i*10] = {cx, cy, a, b, c, sqrt(det), w/2, h/2, cos, sin}
// ---------------------------------------------------------------------------
__global__ void k_gt_prep(const float* __restrict__ gtb,
                          float* __restrict__ gtp, int total) {
  int i = blockIdx.x * blockDim.x + threadIdx.x;
  if (i >= total) return;
  const float* g = gtb + (size_t)i * 5;
  float cx = g[0], cy = g[1], w = g[2], h = g[3], ang = g[4];
  float cs = cosf(ang), sn = sinf(ang);
  float w2 = w * w * (1.0f / 12.0f), h2 = h * h * (1.0f / 12.0f);
  float a = w2 * cs * cs + h2 * sn * sn;
  float b = w2 * sn * sn + h2 * cs * cs;
  float c = (w2 - h2) * cs * sn;
  float det = fmaxf(a * b - c * c, 0.0f);
  float* o = gtp + (size_t)i * 10;
  o[0] = cx; o[1] = cy; o[2] = a; o[3] = b; o[4] = c;
  o[5] = sqrtf(det); o[6] = w * 0.5f; o[7] = h * 0.5f; o[8] = cs; o[9] = sn;
}

// ---------------------------------------------------------------------------
// K0b: per-anchor-box precompute: prp[i*6] = {cx, cy, a, b, c, sqrt(det)}
// ---------------------------------------------------------------------------
__global__ void k_pr_prep(const float* __restrict__ prb,
                          float* __restrict__ prp, int total) {
  int i = blockIdx.x * blockDim.x + threadIdx.x;
  if (i >= total) return;
  const float* g = prb + (size_t)i * 5;
  float cx = g[0], cy = g[1], w = g[2], h = g[3], ang = g[4];
  float cs = cosf(ang), sn = sinf(ang);
  float w2 = w * w * (1.0f / 12.0f), h2 = h * h * (1.0f / 12.0f);
  float a = w2 * cs * cs + h2 * sn * sn;
  float b = w2 * sn * sn + h2 * cs * cs;
  float c = (w2 - h2) * cs * sn;
  float det = fmaxf(a * b - c * c, 0.0f);
  float* o = prp + (size_t)i * 6;
  o[0] = cx; o[1] = cy; o[2] = a; o[3] = b; o[4] = c; o[5] = sqrtf(det);
}

// ---------------------------------------------------------------------------
// K1: fused WMMA class-gather GEMM + ProbIoU + spatial test.
// One wave per 16-anchor tile. cls_scores = one_hot(labels) @ scores^T with
// v_wmma_f32_16x16x32_f16. One-hot A fragments (per (nt,chunk,lane)) are
// built ONCE per block into LDS and re-read via ds_load_b128 by all 8 waves.
// Writes align (unmasked), iou, spatial(byte)  -- each (B,N,L).
// ---------------------------------------------------------------------------
__global__ void k_align_iou(const float* __restrict__ scores,
                            const float* __restrict__ prp,
                            const float* __restrict__ gtp,
                            const int* __restrict__ labels,
                            const float* __restrict__ anch,
                            float* __restrict__ alignA,
                            float* __restrict__ iouB,
                            unsigned char* __restrict__ spatB,
                            int B, int L, int N, int C) {
  __shared__ float sg[64 * 10];
  __shared__ int   slab[64];
  // A fragments: up to 4 nt * 3 chunks * 32 lanes * 8 dwords (packed f16x2)
  __shared__ __align__(32) unsigned int sA[4 * 3 * 32 * 8];

  const int b   = blockIdx.y;
  const int tid = threadIdx.x;
  const int ntc    = (N + 15) >> 4;   // GT tiles (<=4)
  const int chunks = (C + 31) >> 5;   // K chunks of 32 (3 for C=80)

  for (int i = tid; i < N * 10; i += blockDim.x)
    sg[i] = gtp[(size_t)b * N * 10 + i];
  for (int i = tid; i < N; i += blockDim.x)
    slab[i] = labels[b * N + i];
  __syncthreads();

  // Cooperative one-hot A-fragment build (contiguous K order, same as B fragment:
  // lane ln holds row m=ln&15, K = h + 16*(ln>=16)).
  for (int f = tid; f < ntc * chunks * 32; f += blockDim.x) {
    const int ln    = f & 31;
    const int fc    = f >> 5;           // nt*chunks + chunk
    const int nt    = fc / chunks;
    const int chunk = fc - nt * chunks;
    const int lnHi  = ln >> 4;
    const int lab   = slab[nt * 16 + (ln & 15)];
    unsigned int* dst = &sA[(size_t)fc * 32 * 8 + (size_t)ln * 8];
#pragma unroll
    for (int d = 0; d < 8; ++d) {
      const int h0 = 2 * d, h1 = 2 * d + 1;
      const int k0 = chunk * 32 + h0 + (lnHi ? 16 : 0);
      const int k1 = chunk * 32 + h1 + (lnHi ? 16 : 0);
      unsigned int w = 0;
      if (k0 == lab) w |= 0x00003C00u;  // f16 1.0 in low half
      if (k1 == lab) w |= 0x3C000000u;  // f16 1.0 in high half
      dst[d] = w;
    }
  }
  __syncthreads();

  const int wave   = tid >> 5;
  const int lane   = tid & 31;
  const int laneHi = lane >> 4;   // 0 or 1
  const int lane15 = lane & 15;
  const int ltile  = blockIdx.x * (blockDim.x >> 5) + wave;
  if (ltile * 16 >= L) return;            // uniform per wave: EXEC stays all-1s
  const int lr = ltile * 16 + lane15;
  const int l  = (lr < L) ? lr : (L - 1);

  const float* pp = prp + ((size_t)b * L + l) * 6;
  const float x2 = pp[0], y2 = pp[1], a2 = pp[2], b2 = pp[3], c2 = pp[4], sq2 = pp[5];
  const float px = anch[(size_t)l * 2 + 0], py = anch[(size_t)l * 2 + 1];

  v8f acc[4];
#pragma unroll
  for (int t = 0; t < 4; ++t)
#pragma unroll
    for (int r = 0; r < 8; ++r) acc[t][r] = 0.0f;

  const float* srow = scores + ((size_t)b * L + l) * (size_t)C;
  for (int chunk = 0; chunk < chunks; ++chunk) {
    // B fragment: 32x16 f16, lane holds col n = lane&15, K = 16*laneHi + h
    v16h bf;
    const int cbase = chunk * 32 + laneHi * 16;
    if (cbase + 16 <= C) {
      const float4* p4 = (const float4*)(srow + cbase);  // 16B aligned
#pragma unroll
      for (int q = 0; q < 4; ++q) {
        float4 t = p4[q];
        bf[q * 4 + 0] = (_Float16)t.x;
        bf[q * 4 + 1] = (_Float16)t.y;
        bf[q * 4 + 2] = (_Float16)t.z;
        bf[q * 4 + 3] = (_Float16)t.w;
      }
    } else {
#pragma unroll
      for (int hh = 0; hh < 16; ++hh) {
        int c = cbase + hh;
        bf[hh] = (c < C) ? (_Float16)srow[c] : (_Float16)0.0f;
      }
    }
#pragma unroll
    for (int nt = 0; nt < 4; ++nt) {
      if (nt >= ntc) break;
      const v16h af = *reinterpret_cast<const v16h*>(
          &sA[((size_t)(nt * chunks + chunk) * 32 + lane) * 8]);
      acc[nt] = __builtin_amdgcn_wmma_f32_16x16x32_f16(
          false, af, false, bf, (short)0, acc[nt], false, false);
    }
  }

  // Consume accumulator: C layout => lane holds (M = r + 8*laneHi, N = lane&15)
#pragma unroll
  for (int nt = 0; nt < 4; ++nt) {
    if (nt >= ntc) break;
#pragma unroll
    for (int r = 0; r < 8; ++r) {
      const int n = nt * 16 + laneHi * 8 + r;
      if (n >= N) break;
      const float* g = &sg[n * 10];
      const float x1 = g[0], y1 = g[1], a1 = g[2], b1 = g[3], c1 = g[4];
      const float sq1 = g[5], hw = g[6], hh2 = g[7], ca = g[8], sa = g[9];
      // ProbIoU
      float dx = x1 - x2, dy = y1 - y2;
      float As = a1 + a2, Bs = b1 + b2, Cs = c1 + c2;
      float denom = As * Bs - Cs * Cs + IOU_EPSF;
      float inv = 1.0f / denom;
      float t1 = 0.25f * (As * dy * dy + Bs * dx * dx) * inv;
      float t2 = 0.5f * Cs * (-dx) * dy * inv;
      float t3 = 0.5f * logf(denom / (4.0f * sq1 * sq2 + IOU_EPSF) + IOU_EPSF);
      float bd = fminf(fmaxf(t1 + t2 + t3, IOU_EPSF), 100.0f);
      float hd = sqrtf(1.0f - expf(-bd) + IOU_EPSF);
      float iou = fminf(fmaxf(1.0f - hd, 0.0f), 1.0f);
      // point-in-rotated-box
      float pdx = px - x1, pdy = py - y1;
      float lx = pdx * ca + pdy * sa;
      float ly = -pdx * sa + pdy * ca;
      bool sp = (fabsf(lx) <= hw) && (fabsf(ly) <= hh2);
      // align = cls^1 * iou^6
      float cls = acc[nt][r];
      float i2 = iou * iou;
      float algn = cls * (i2 * i2 * i2);
      if (lr < L) {
        size_t o = (size_t)(b * N + n) * (size_t)L + (size_t)lr;
        alignA[o] = algn;
        iouB[o]   = iou;
        spatB[o]  = sp ? (unsigned char)1 : (unsigned char)0;
      }
    }
  }
}

// ---------------------------------------------------------------------------
// K2: k-th largest masked value per (b,n) row via 13 strict-descending maxes.
// kth==0 means "< k positives; select all masked>0".
// ---------------------------------------------------------------------------
__global__ void k_topk_thresh(const float* __restrict__ alignA,
                              const unsigned char* __restrict__ spatB,
                              float* __restrict__ kth, int L) {
  __shared__ float red[256];
  const int bn = blockIdx.x;
  const float* row = alignA + (size_t)bn * L;
  const unsigned char* srow = spatB + (size_t)bn * L;
  const int tid = threadIdx.x;
  float thresh = 3.4e38f;
  for (int it = 0; it < TOPK_K; ++it) {
    float loc = -1.0f;
    for (int i = tid; i < L; i += 256) {
      if (i + 1024 < L) __builtin_prefetch(&row[i + 1024], 0, 0);
      float v = srow[i] ? row[i] : 0.0f;
      if (v < thresh && v > loc) loc = v;
    }
    red[tid] = loc;
    __syncthreads();
    for (int s = 128; s > 0; s >>= 1) {
      if (tid < s) red[tid] = fmaxf(red[tid], red[tid + s]);
      __syncthreads();
    }
    float m = red[0];
    __syncthreads();
    if (m <= 0.0f) { thresh = 0.0f; break; }
    thresh = m;
  }
  if (tid == 0) kth[bn] = (thresh > 3.0e38f) ? 0.0f : thresh;
}

// ---------------------------------------------------------------------------
// K3: per-anchor: build positive bitmask, conflict resolution via max-IoU,
// assigned labels / bboxes, store bitmask + class index.
// ---------------------------------------------------------------------------
__global__ void k_resolve(const float* __restrict__ alignA,
                          const unsigned char* __restrict__ spatB,
                          const float* __restrict__ iouB,
                          const float* __restrict__ kth,
                          const float* __restrict__ pad,
                          const int* __restrict__ labels,
                          const float* __restrict__ gtb,
                          const int* __restrict__ bgp,
                          unsigned long long* __restrict__ posbits,
                          int* __restrict__ clsI,
                          float* __restrict__ outL,
                          float* __restrict__ outBox,
                          int B, int L, int N) {
  const int b = blockIdx.y;
  const int l = blockIdx.x * blockDim.x + threadIdx.x;
  if (l >= L) return;
  unsigned long long bits = 0ull;
  int ps = 0;
  float ioumax = -1.0f;
  for (int n = 0; n < N; ++n) {
    size_t o = (size_t)(b * N + n) * (size_t)L + (size_t)l;
    float ma = spatB[o] ? alignA[o] : 0.0f;
    float iv = iouB[o];
    float th = kth[b * N + n];
    bool bit = (pad[b * N + n] > 0.0f) &&
               ((th > 0.0f) ? (ma >= th) : (ma > 0.0f));
    if (bit) { bits |= (1ull << n); ++ps; }
    ioumax = fmaxf(ioumax, iv);
  }
  if (ps > 1) {  // keep GT(s) with max IoU (spatial/pad NOT applied, per ref)
    bits = 0ull;
    for (int n = 0; n < N; ++n) {
      size_t o = (size_t)(b * N + n) * (size_t)L + (size_t)l;
      if (iouB[o] == ioumax) bits |= (1ull << n);
    }
  }
  const int  gt  = bits ? (int)__builtin_ctzll(bits) : 0;  // argmax = first set
  const bool fg  = (bits != 0ull);
  const int  lab = fg ? labels[b * N + gt] : bgp[0];
  const size_t bl = (size_t)b * L + (size_t)l;
  outL[bl] = (float)lab;
  for (int j = 0; j < 5; ++j)
    outBox[bl * 5 + j] = fg ? gtb[(size_t)(b * N + gt) * 5 + j] : 0.0f;
  posbits[bl] = bits;
  clsI[bl] = lab;
}

// ---------------------------------------------------------------------------
// K4: per-GT maxima over anchors of align*positive and iou*positive.
// ---------------------------------------------------------------------------
__global__ void k_gt_max(const float* __restrict__ alignA,
                         const float* __restrict__ iouB,
                         const unsigned long long* __restrict__ posbits,
                         float* __restrict__ maxam,
                         float* __restrict__ maxip, int L, int N) {
  __shared__ float r0[256], r1[256];
  const int bn = blockIdx.x;
  const int b = bn / N, n = bn % N;
  const float* arow = alignA + (size_t)bn * L;
  const float* irow = iouB + (size_t)bn * L;
  const unsigned long long* pb = posbits + (size_t)b * L;
  float mam = 0.0f, mip = 0.0f;
  for (int i = threadIdx.x; i < L; i += 256) {
    if (i + 1024 < L) __builtin_prefetch(&pb[i + 1024], 0, 0);
    if ((pb[i] >> n) & 1ull) {
      mam = fmaxf(mam, arow[i]);
      mip = fmaxf(mip, irow[i]);
    }
  }
  r0[threadIdx.x] = mam; r1[threadIdx.x] = mip;
  __syncthreads();
  for (int s = 128; s > 0; s >>= 1) {
    if (threadIdx.x < s) {
      r0[threadIdx.x] = fmaxf(r0[threadIdx.x], r0[threadIdx.x + s]);
      r1[threadIdx.x] = fmaxf(r1[threadIdx.x], r1[threadIdx.x + s]);
    }
    __syncthreads();
  }
  if (threadIdx.x == 0) { maxam[bn] = r0[0]; maxip[bn] = r1[0]; }
}

// ---------------------------------------------------------------------------
// K5a: per-anchor score normalization factor.
// ---------------------------------------------------------------------------
__global__ void k_norm(const float* __restrict__ alignA,
                       const unsigned long long* __restrict__ posbits,
                       const float* __restrict__ maxam,
                       const float* __restrict__ maxip,
                       float* __restrict__ normW, int L, int N) {
  const int b = blockIdx.y;
  const int l = blockIdx.x * blockDim.x + threadIdx.x;
  if (l >= L) return;
  unsigned long long bits = posbits[(size_t)b * L + l];
  float nrm = 0.0f;
  while (bits) {
    int n = (int)__builtin_ctzll(bits);
    bits &= bits - 1ull;
    size_t o = (size_t)(b * N + n) * (size_t)L + (size_t)l;
    float v = alignA[o] / (maxam[b * N + n] + NORM_EPS) * maxip[b * N + n];
    nrm = fmaxf(nrm, v);
  }
  normW[(size_t)b * L + l] = nrm;
}

// ---------------------------------------------------------------------------
// K5b: scatter one-hot * norm into assigned_scores (B,L,C), coalesced.
// ---------------------------------------------------------------------------
__global__ void k_scatter(const float* __restrict__ normW,
                          const int* __restrict__ clsI,
                          float* __restrict__ outS,
                          unsigned long long total, int C) {
  unsigned long long i =
      (unsigned long long)blockIdx.x * blockDim.x + threadIdx.x;
  if (i >= total) return;
  unsigned long long bl = i / (unsigned int)C;
  int c = (int)(i - bl * (unsigned int)C);
  outS[i] = (c == clsI[bl]) ? normW[bl] : 0.0f;
}

// ---------------------------------------------------------------------------
extern "C" void kernel_launch(void* const* d_in, const int* in_sizes, int n_in,
                              void* d_out, int out_size, void* d_ws, size_t ws_size,
                              hipStream_t stream) {
  const float* scores = (const float*)d_in[0];
  const float* prb    = (const float*)d_in[1];
  const float* anch   = (const float*)d_in[2];
  const int*   labels = (const int*)d_in[3];
  const float* gtb    = (const float*)d_in[4];
  const float* pad    = (const float*)d_in[5];
  const int*   bgp    = (const int*)d_in[6];

  const int L = in_sizes[2] / 2;
  const int B = in_sizes[1] / (5 * L);
  const int N = in_sizes[3] / B;
  const int C = in_sizes[0] / (B * L);

  size_t off = 0;
  auto take = [&](size_t bytes) -> void* {
    off = (off + 255) & ~(size_t)255;
    void* p = (char*)d_ws + off;
    off += bytes;
    return p;
  };
  float*              gtp     = (float*)take((size_t)B * N * 10 * 4);
  float*              prp     = (float*)take((size_t)B * L * 6 * 4);
  float*              alignA  = (float*)take((size_t)B * N * L * 4);
  float*              iouB    = (float*)take((size_t)B * N * L * 4);
  unsigned char*      spatB   = (unsigned char*)take((size_t)B * N * L);
  float*              kth     = (float*)take((size_t)B * N * 4);
  unsigned long long* posbits = (unsigned long long*)take((size_t)B * L * 8);
  float*              maxam   = (float*)take((size_t)B * N * 4);
  float*              maxip   = (float*)take((size_t)B * N * 4);
  float*              normW   = (float*)take((size_t)B * L * 4);
  int*                clsI    = (int*)take((size_t)B * L * 4);

  float* outL   = (float*)d_out;                     // (B,L)   labels as float
  float* outBox = outL + (size_t)B * L;              // (B,L,5)
  float* outS   = outBox + (size_t)B * L * 5;        // (B,L,C)

  k_gt_prep<<<dim3((B * N + 255) / 256), dim3(256), 0, stream>>>(gtb, gtp, B * N);
  k_pr_prep<<<dim3((unsigned)(((size_t)B * L + 255) / 256)), dim3(256), 0, stream>>>(
      prb, prp, B * L);

  const int tiles = (L + 15) / 16;
  k_align_iou<<<dim3((tiles + 7) / 8, B), dim3(256), 0, stream>>>(
      scores, prp, gtp, labels, anch, alignA, iouB, spatB, B, L, N, C);

  k_topk_thresh<<<dim3(B * N), dim3(256), 0, stream>>>(alignA, spatB, kth, L);

  k_resolve<<<dim3((L + 255) / 256, B), dim3(256), 0, stream>>>(
      alignA, spatB, iouB, kth, pad, labels, gtb, bgp,
      posbits, clsI, outL, outBox, B, L, N);

  k_gt_max<<<dim3(B * N), dim3(256), 0, stream>>>(alignA, iouB, posbits,
                                                  maxam, maxip, L, N);

  k_norm<<<dim3((L + 255) / 256, B), dim3(256), 0, stream>>>(
      alignA, posbits, maxam, maxip, normW, L, N);

  const unsigned long long totalS = (unsigned long long)B * L * C;
  k_scatter<<<dim3((unsigned)((totalS + 255) / 256)), dim3(256), 0, stream>>>(
      normW, clsI, outS, totalS, C);
}
